// Attention_11098195493651
// MI455X (gfx1250) — compile-verified, matrix operand order checked by reading, not run on hardware
//
#include <hip/hip_runtime.h>

typedef __attribute__((ext_vector_type(16))) __bf16 v16bf;
typedef __attribute__((ext_vector_type(8)))  __bf16 v8bf;
typedef __attribute__((ext_vector_type(8)))  float  v8f;
typedef int v4i __attribute__((vector_size(16)));

#define T_LEN 4096
#define C_DIM 2048
#define NH    16
#define NKV   4
#define HD    128
#define WIN   1024
#define NB    4

// gfx1250 async global->LDS path (guarded; host pass / older toolchains take
// the synchronous staging fallback automatically).
#if defined(__gfx1250__) && \
    __has_builtin(__builtin_amdgcn_global_load_async_to_lds_b128) && \
    __has_builtin(__builtin_amdgcn_s_wait_asynccnt)
#define HAVE_ASYNC_LDS 1
#else
#define HAVE_ASYNC_LDS 0
#endif

#define AS_GLOBAL __attribute__((address_space(1)))
#define AS_LDS    __attribute__((address_space(3)))

static __device__ __forceinline__ v8f vzero() {
  v8f z;
#pragma unroll
  for (int i = 0; i < 8; ++i) z[i] = 0.0f;
  return z;
}

// Loads a 16x32 bf16 operand fragment slice for one lane.
// Per CDNA5 ISA: lanes 0-15 hold K={0..7,16..23}, lanes 16-31 hold K={8..15,24..31}
// relative to the chunk base; caller passes (row_ptr + chunk + half*8).
static __device__ __forceinline__ v16bf load_frag_g(const __bf16* p) {
  union { v16bf v; v8bf h[2]; } u;
  u.h[0] = *(const v8bf*)(p);
  u.h[1] = *(const v8bf*)(p + 16);
  return u.v;
}

static __device__ __forceinline__ v8f wmma_bf16(v16bf a, v16bf b, v8f c) {
  return __builtin_amdgcn_wmma_f32_16x16x32_bf16(false, a, false, b, (short)0, c,
                                                 false, false);
}

// ---------------- conversion kernels ----------------

__global__ void cvt_bf16_kernel(const float* __restrict__ in,
                                __bf16* __restrict__ out, int n) {
  int i = blockIdx.x * blockDim.x + threadIdx.x;
  if (i < n) out[i] = (__bf16)in[i];
}

// in [R][C] fp32 -> out [C][R] bf16 (transposed)
__global__ void cvt_transpose_kernel(const float* __restrict__ in,
                                     __bf16* __restrict__ out, int R, int C) {
  size_t i = (size_t)blockIdx.x * blockDim.x + threadIdx.x;
  if (i >= (size_t)R * C) return;
  int r = (int)(i / C), c = (int)(i % C);
  out[(size_t)c * R + r] = (__bf16)in[i];
}

// in [T][Ncols] bf16 -> out [Ncols][T] bf16
__global__ void transpose_bf16_kernel(const __bf16* __restrict__ in,
                                      __bf16* __restrict__ out, int R, int C) {
  size_t i = (size_t)blockIdx.x * blockDim.x + threadIdx.x;
  if (i >= (size_t)R * C) return;
  int r = (int)(i / C), c = (int)(i % C);
  out[(size_t)c * R + r] = in[i];
}

// ---------------- RoPE (in place on bf16 [T][nheads*128]) ----------------

__global__ void rope_kernel(__bf16* __restrict__ u, int nheads) {
  int idx = blockIdx.x * blockDim.x + threadIdx.x;
  int total = T_LEN * nheads * 64;
  if (idx >= total) return;
  int i = idx & 63;
  int h = (idx >> 6) % nheads;
  int t = idx / (64 * nheads);
  float ts  = powf(10000.0f, (float)i * (1.0f / 64.0f));
  float ang = (float)t / ts;
  float s = __sinf(ang), c = __cosf(ang);
  __bf16* p = u + (size_t)t * nheads * HD + h * HD;
  float x1 = (float)p[i];
  float x2 = (float)p[i + 64];
  p[i]      = (__bf16)(x1 * c - x2 * s);
  p[i + 64] = (__bf16)(x2 * c + x1 * s);
}

// ---------------- bf16 WMMA GEMM: C[M][N] = A[M][K] * BT[N][K]^T ----------------
// 4 waves (128 thr) per block, 64x64 block tile, wave computes 32x32 (2x2 frags).
// A/B 64x32 tiles double-buffered in LDS, filled with async global->LDS copies.
#define LPAD 40

__global__ __launch_bounds__(128) void gemm_bf16_kernel(
    const __bf16* __restrict__ A, const __bf16* __restrict__ BT,
    __bf16* __restrict__ outH, float* __restrict__ outF, int M, int N, int K) {
  __shared__ __align__(16) __bf16 Asm[2][64][LPAD];
  __shared__ __align__(16) __bf16 Bsm[2][64][LPAD];

  const int tid = threadIdx.x;
  const int wave = tid >> 5;
  const int lane = tid & 31;
  const int l16 = lane & 15;
  const int hl  = lane >> 4;
  const int mblk = blockIdx.x * 64;
  const int nblk = blockIdx.y * 64;
  const int ms = (wave & 1) * 32;
  const int ns = (wave >> 1) * 32;

  // Stage one 64x32 A tile + one 64x32 B tile into LDS buffer `buf`.
  // Each thread moves 2x16B of A and 2x16B of B (coalesced).
  auto stage = [&](int buf, int k0) {
#pragma unroll
    for (int qq = 0; qq < 2; ++qq) {
      int lin = qq * 128 + tid;     // 0..255
      int row = lin >> 2;           // 0..63
      int col = (lin & 3) * 8;      // 0,8,16,24 (bf16 elems)
      const __bf16* ga = A  + (size_t)(mblk + row) * K + k0 + col;
      const __bf16* gb = BT + (size_t)(nblk + row) * K + k0 + col;
#if HAVE_ASYNC_LDS
      __builtin_amdgcn_global_load_async_to_lds_b128(
          (AS_GLOBAL v4i*)ga, (AS_LDS v4i*)&Asm[buf][row][col], 0, 0);
      __builtin_amdgcn_global_load_async_to_lds_b128(
          (AS_GLOBAL v4i*)gb, (AS_LDS v4i*)&Bsm[buf][row][col], 0, 0);
#else
      *(v8bf*)&Asm[buf][row][col] = *(const v8bf*)ga;
      *(v8bf*)&Bsm[buf][row][col] = *(const v8bf*)gb;
#endif
    }
  };

  v8f acc[2][2];
#pragma unroll
  for (int i = 0; i < 2; ++i)
#pragma unroll
    for (int j = 0; j < 2; ++j) acc[i][j] = vzero();

  const int nk = K / 32;
  stage(0, 0);

  for (int ck = 0; ck < nk; ++ck) {
    const int buf = ck & 1;
    if (ck + 1 < nk) {
      stage(buf ^ 1, (ck + 1) * 32);
#if HAVE_ASYNC_LDS
      __builtin_amdgcn_s_wait_asynccnt(4);  // this chunk's 4 copies landed
#endif
    } else {
#if HAVE_ASYNC_LDS
      __builtin_amdgcn_s_wait_asynccnt(0);
#endif
    }
    __syncthreads();  // all waves' copies for `buf` visible

    v16bf a0 = load_frag_g(&Asm[buf][ms + l16][hl * 8]);
    v16bf a1 = load_frag_g(&Asm[buf][ms + 16 + l16][hl * 8]);
    v16bf b0 = load_frag_g(&Bsm[buf][ns + l16][hl * 8]);
    v16bf b1 = load_frag_g(&Bsm[buf][ns + 16 + l16][hl * 8]);
    acc[0][0] = wmma_bf16(a0, b0, acc[0][0]);
    acc[0][1] = wmma_bf16(a0, b1, acc[0][1]);
    acc[1][0] = wmma_bf16(a1, b0, acc[1][0]);
    acc[1][1] = wmma_bf16(a1, b1, acc[1][1]);

    __syncthreads();  // done reading `buf` before it gets overwritten
  }

#pragma unroll
  for (int i = 0; i < 2; ++i)
#pragma unroll
    for (int j = 0; j < 2; ++j)
#pragma unroll
      for (int g = 0; g < 8; ++g) {
        int row = mblk + ms + i * 16 + hl * 8 + g;
        int col = nblk + ns + j * 16 + l16;
        if (outF) outF[(size_t)row * N + col] = acc[i][j][g];
        else      outH[(size_t)row * N + col] = (__bf16)acc[i][j][g];
      }
}

// ---------------- sliding-window block attention ----------------
// grid: (8 q-tiles of 128, NH heads, NB blocks); block = 256 threads (8 waves).
// Each wave handles 16 query rows, online softmax over 32-key chunks.
__global__ __launch_bounds__(256) void attn_kernel(
    const __bf16* __restrict__ q,   // [T][NH*HD], RoPE'd
    const __bf16* __restrict__ k,   // [T][NKV*HD], RoPE'd
    const __bf16* __restrict__ vT,  // [NKV*HD][T]
    __bf16* __restrict__ o) {       // [T][NH*HD]
  __shared__ __align__(16) __bf16 plds[8][16][40];

  const int wave = threadIdx.x >> 5;
  const int lane = threadIdx.x & 31;
  const int l16 = lane & 15;
  const int hl  = lane >> 4;
  const int qt = blockIdx.x;   // 0..7
  const int h  = blockIdx.y;   // 0..15
  const int n  = blockIdx.z;   // 0..3
  const int kh = h >> 2;       // GQA: 4 query heads per kv head
  const int t0 = n * WIN + qt * 128 + wave * 16;
  const int p0 = qt * 128;

  // Q fragments (16 rows x 128 d, as 4 chunks of K=32)
  const __bf16* qrow = q + (size_t)(t0 + l16) * (NH * HD) + h * HD;
  v16bf qa[4];
#pragma unroll
  for (int c = 0; c < 4; ++c) qa[c] = load_frag_g(qrow + c * 32 + hl * 8);

  v8f accO[8];
  float mrow[8], lsum[8];
#pragma unroll
  for (int dt = 0; dt < 8; ++dt) accO[dt] = vzero();
#pragma unroll
  for (int g = 0; g < 8; ++g) { mrow[g] = -1e30f; lsum[g] = 0.0f; }

  // block-uniform chunk range over j in [0, 2W)
  int jmin = (n == 0) ? ((p0 > WIN) ? p0 : WIN) : p0;
  int c_lo = jmin >> 5;
  int c_hi = (p0 + 127 + WIN) >> 5;
  if (c_hi > (2 * WIN) / 32 - 1) c_hi = (2 * WIN) / 32 - 1;
  const float scale = 0.08838834764831845f;  // 1/sqrt(128)

  for (int ch = c_lo; ch <= c_hi; ++ch) {
    const int j0  = ch << 5;
    const int tkb = n * WIN - WIN + j0;  // >= 0 within the chunk range

    // S = Q K^T for 16 rows x 32 keys (two 16x16 tiles)
    v8f S[2];
    S[0] = vzero(); S[1] = vzero();
#pragma unroll
    for (int s = 0; s < 2; ++s) {
      const __bf16* krow = k + (size_t)(tkb + s * 16 + l16) * (NKV * HD) + kh * HD;
#pragma unroll
      for (int c = 0; c < 4; ++c) {
        v16bf kb = load_frag_g(krow + c * 32 + hl * 8);
        S[s] = wmma_bf16(qa[c], kb, S[s]);
      }
    }

    // scale + mask
    float sv[2][8];
#pragma unroll
    for (int s = 0; s < 2; ++s) {
      const int j = j0 + s * 16 + l16;
#pragma unroll
      for (int g = 0; g < 8; ++g) {
        const int p = p0 + wave * 16 + hl * 8 + g;
        bool ok = (j >= p) && (j <= p + WIN) && ((n > 0) || (j >= WIN));
        sv[s][g] = ok ? S[s][g] * scale : -__builtin_inff();
      }
    }

    // row max over the 32 keys (reduce within 16-lane halves; rows 0-7 live in
    // lanes 0-15, rows 8-15 in lanes 16-31, matching the C-fragment layout)
    float mt[8];
#pragma unroll
    for (int g = 0; g < 8; ++g) mt[g] = fmaxf(sv[0][g], sv[1][g]);
#pragma unroll
    for (int off = 1; off < 16; off <<= 1)
#pragma unroll
      for (int g = 0; g < 8; ++g) mt[g] = fmaxf(mt[g], __shfl_xor(mt[g], off, 32));

    float alpha[8];
#pragma unroll
    for (int g = 0; g < 8; ++g) {
      float mn = fmaxf(mrow[g], mt[g]);
      alpha[g] = __expf(mrow[g] - mn);
      mrow[g] = mn;
    }
#pragma unroll
    for (int s = 0; s < 2; ++s)
#pragma unroll
      for (int g = 0; g < 8; ++g) sv[s][g] = __expf(sv[s][g] - mrow[g]);

    float rs[8];
#pragma unroll
    for (int g = 0; g < 8; ++g) rs[g] = sv[0][g] + sv[1][g];
#pragma unroll
    for (int off = 1; off < 16; off <<= 1)
#pragma unroll
      for (int g = 0; g < 8; ++g) rs[g] += __shfl_xor(rs[g], off, 32);
#pragma unroll
    for (int g = 0; g < 8; ++g) lsum[g] = lsum[g] * alpha[g] + rs[g];
#pragma unroll
    for (int dt = 0; dt < 8; ++dt)
#pragma unroll
      for (int g = 0; g < 8; ++g) accO[dt][g] *= alpha[g];

    // P (C-layout) -> LDS -> A-fragment layout (per-wave region; DS ops from a
    // wave are in-order, wave_barrier stops compiler reordering)
#pragma unroll
    for (int s = 0; s < 2; ++s)
#pragma unroll
      for (int g = 0; g < 8; ++g)
        plds[wave][hl * 8 + g][s * 16 + l16] = (__bf16)sv[s][g];
    __builtin_amdgcn_wave_barrier();
    union { v16bf v; v8bf hh[2]; } pu;
    pu.hh[0] = *(const v8bf*)&plds[wave][l16][hl * 8];
    pu.hh[1] = *(const v8bf*)&plds[wave][l16][hl * 8 + 16];
    __builtin_amdgcn_wave_barrier();

    // O += P * V  (V^T rows are contiguous along t)
#pragma unroll
    for (int dt = 0; dt < 8; ++dt) {
      const __bf16* vrow =
          vT + (size_t)(kh * HD + dt * 16 + l16) * T_LEN + tkb + hl * 8;
      v16bf vb = load_frag_g(vrow);
      accO[dt] = wmma_bf16(pu.v, vb, accO[dt]);
    }
  }

  // normalize + store
#pragma unroll
  for (int g = 0; g < 8; ++g) {
    const float inv = 1.0f / lsum[g];
    const int trow = t0 + hl * 8 + g;
#pragma unroll
    for (int dt = 0; dt < 8; ++dt)
      o[(size_t)trow * (NH * HD) + h * HD + dt * 16 + l16] =
          (__bf16)(accO[dt][g] * inv);
  }
}

// ---------------- host launch ----------------

extern "C" void kernel_launch(void* const* d_in, const int* in_sizes, int n_in,
                              void* d_out, int out_size, void* d_ws, size_t ws_size,
                              hipStream_t stream) {
  (void)in_sizes; (void)n_in; (void)out_size; (void)ws_size;
  const float* x  = (const float*)d_in[0];
  const float* wq = (const float*)d_in[1];
  const float* wk = (const float*)d_in[2];
  const float* wv = (const float*)d_in[3];
  const float* wo = (const float*)d_in[4];
  float* out = (float*)d_out;

  char* ws = (char*)d_ws;
  __bf16* xh   = (__bf16*)(ws + 0);          // [4096][2048]     16 MB
  __bf16* wqT  = (__bf16*)(ws + 16777216);   // [2048][2048]      8 MB
  __bf16* wkT  = (__bf16*)(ws + 25165824);   // [512][2048]       2 MB
  __bf16* wvT  = (__bf16*)(ws + 27262976);   // [512][2048]       2 MB
  __bf16* woT  = (__bf16*)(ws + 29360128);   // [2048][2048]      8 MB
  __bf16* qh   = (__bf16*)(ws + 37748736);   // [4096][2048]     16 MB
  __bf16* khb  = (__bf16*)(ws + 54525952);   // [4096][512]       4 MB
  __bf16* vhb  = (__bf16*)(ws + 58720256);   // [4096][512]       4 MB
  __bf16* vTb  = (__bf16*)(ws + 62914560);   // [512][4096]       4 MB
  __bf16* attn = (__bf16*)(ws + 67108864);   // [4096][2048]     16 MB

  // fp32 -> bf16 conversions (weights transposed for contiguous B-fragments)
  {
    int n = T_LEN * C_DIM;
    cvt_bf16_kernel<<<(n + 255) / 256, 256, 0, stream>>>(x, xh, n);
  }
  {
    int n = C_DIM * (NH * HD);
    cvt_transpose_kernel<<<(n + 255) / 256, 256, 0, stream>>>(wq, wqT, C_DIM, NH * HD);
  }
  {
    int n = C_DIM * (NKV * HD);
    cvt_transpose_kernel<<<(n + 255) / 256, 256, 0, stream>>>(wk, wkT, C_DIM, NKV * HD);
    cvt_transpose_kernel<<<(n + 255) / 256, 256, 0, stream>>>(wv, wvT, C_DIM, NKV * HD);
  }
  {
    int n = (NH * HD) * C_DIM;
    cvt_transpose_kernel<<<(n + 255) / 256, 256, 0, stream>>>(wo, woT, NH * HD, C_DIM);
  }

  // QKV projections
  gemm_bf16_kernel<<<dim3(T_LEN / 64, (NH * HD) / 64), 128, 0, stream>>>(
      xh, wqT, qh, nullptr, T_LEN, NH * HD, C_DIM);
  gemm_bf16_kernel<<<dim3(T_LEN / 64, (NKV * HD) / 64), 128, 0, stream>>>(
      xh, wkT, khb, nullptr, T_LEN, NKV * HD, C_DIM);
  gemm_bf16_kernel<<<dim3(T_LEN / 64, (NKV * HD) / 64), 128, 0, stream>>>(
      xh, wvT, vhb, nullptr, T_LEN, NKV * HD, C_DIM);

  // RoPE on q and k
  {
    int n = T_LEN * NH * 64;
    rope_kernel<<<(n + 255) / 256, 256, 0, stream>>>(qh, NH);
  }
  {
    int n = T_LEN * NKV * 64;
    rope_kernel<<<(n + 255) / 256, 256, 0, stream>>>(khb, NKV);
  }

  // V transpose for contiguous PV B-fragments
  {
    int n = T_LEN * NKV * HD;
    transpose_bf16_kernel<<<(n + 255) / 256, 256, 0, stream>>>(vhb, vTb, T_LEN, NKV * HD);
  }

  // block-local sliding-window attention
  attn_kernel<<<dim3(8, NH, NB), 256, 0, stream>>>(qh, khb, vTb, attn);

  // output projection -> fp32 d_out
  gemm_bf16_kernel<<<dim3(T_LEN / 64, C_DIM / 64), 128, 0, stream>>>(
      attn, woT, nullptr, out, T_LEN, C_DIM, C_DIM);
}
